// EquivariantSelfAttention_2379411881985
// MI455X (gfx1250) — compile-verified, matrix operand order checked by reading, not run on hardware
//
#include <hip/hip_runtime.h>
#include <hip/hip_bf16.h>
#include <math.h>

// ---------------------------------------------------------------------------
// GNN edge MLP + scatter for MI455X (gfx1250, wave32, WMMA 16x16x32 bf16)
// ---------------------------------------------------------------------------

typedef __attribute__((ext_vector_type(16))) __bf16 v16bf;
typedef __attribute__((ext_vector_type(8)))  __bf16 v8bf;
typedef __attribute__((ext_vector_type(8)))  float  v8f;

#define NNODES 50000
#define NEDGES 800000
#define HID    128
#define NBAS   32
#define EIN    288          // 2*HID + NBAS
#define NT     8            // 8 N-tiles of 16 cover HID=128
#define KT1    9            // 9 K-tiles of 32 cover EIN=288
#define KT2    4            // 4 K-tiles of 32 cover HID=128
#define CUTOFF_F 5.0f

#define MSG_STRIDE 304      // row stride (bf16 elems) for 16x288 msg tile, 16B-mult
#define H1_STRIDE  136      // row stride (bf16 elems) for 16x128 tiles, 16B-mult

// --- A-fragment load from a row-major bf16 row (ISA 16-bit A layout):
// lane l holds row M=l&15; elements 0..7 = K(col0..col0+7), 8..15 = K(col0+16..+23)
// where col0 = ktile*32 + 8*(l>=16). Two aligned 16B LDS loads.
__device__ __forceinline__ v16bf load_frag_row(const __bf16* __restrict__ row, int col0) {
    v8bf lo = *(const v8bf*)(row + col0);
    v8bf hi = *(const v8bf*)(row + col0 + 16);
    v16bf r;
#pragma unroll
    for (int i = 0; i < 8; ++i) { r[i] = lo[i]; r[i + 8] = hi[i]; }
    return r;
}

// --- B-fragment load from pre-swizzled weight LDS (16 contiguous bf16 per lane)
__device__ __forceinline__ v16bf load_frag_packed(const __bf16* __restrict__ p) {
    v8bf lo = *(const v8bf*)(p);
    v8bf hi = *(const v8bf*)(p + 8);
    v16bf r;
#pragma unroll
    for (int i = 0; i < 8; ++i) { r[i] = lo[i]; r[i + 8] = hi[i]; }
    return r;
}

// Stage a [ktiles*32, 128] fp32 weight matrix into LDS bf16, pre-swizzled into
// B-fragment order: idx = (((kt*8 + nt)*2 + h)*16 + n)*16 + i, where the lane
// (h = lane>=16, n = lane&15) reads 16 contiguous bf16.
// Element i maps to K = kt*32 + (i<8 ? i+8h : i+8+8h), col = nt*16 + n.
__device__ __forceinline__ void stage_weights_frag(const float* __restrict__ W,
                                                   __bf16* __restrict__ dst,
                                                   int ktiles, int tid, int nthreads) {
    int total = ktiles * NT * 2 * 16 * 16;
    for (int idx = tid; idx < total; idx += nthreads) {
        int i  = idx & 15;
        int n  = (idx >> 4) & 15;
        int h  = (idx >> 8) & 1;
        int nt = (idx >> 9) & 7;
        int kt = idx >> 12;
        int krel = (i < 8) ? (i + 8 * h) : (i + 8 + 8 * h);
        int k    = kt * 32 + krel;
        int col  = nt * 16 + n;
        dst[idx] = (__bf16)W[(size_t)k * HID + col];
    }
}

__global__ void zero_kernel(float* __restrict__ p, size_t n) {
    size_t i = (size_t)blockIdx.x * blockDim.x + threadIdx.x;
    size_t stride = (size_t)gridDim.x * blockDim.x;
    for (; i < n; i += stride) p[i] = 0.0f;
}

// ---------------------------------------------------------------------------
// Edge kernel: per 16-edge tile, msg = [x_src | x_dst | rbf] -> silu(msg@W1+b1)@W2+b2
// scatter-add into updates[dst], count degree.  8 waves/block, 1 tile/wave.
// ---------------------------------------------------------------------------
__global__ void __launch_bounds__(256, 1)
edge_mlp_kernel(const float* __restrict__ nf, const float* __restrict__ pos,
                const int* __restrict__ ei,
                const float* __restrict__ W1, const float* __restrict__ b1,
                const float* __restrict__ W2, const float* __restrict__ b2,
                float* __restrict__ updates, float* __restrict__ degree,
                int numTiles) {
    extern __shared__ __align__(16) char smem[];
    __bf16* sW1f  = (__bf16*)smem;                       // 9*4096 elems
    __bf16* sW2f  = sW1f + KT1 * 4096;                   // 4*4096 elems
    __bf16* sMsg  = sW2f + KT2 * 4096;                   // 8*16*MSG_STRIDE
    __bf16* sH1   = sMsg + 8 * 16 * MSG_STRIDE;          // 8*16*H1_STRIDE
    int*    sDst  = (int*)(sH1 + 8 * 16 * H1_STRIDE);    // 8*16
    float*  sDist = (float*)(sDst + 8 * 16);             // 8*16

    const int tid = threadIdx.x;
    const int w   = tid >> 5;      // wave in block
    const int l   = tid & 31;      // lane
    const int nl  = l & 15;        // N column / row-in-tile selector
    const int hl  = l >> 4;        // lane half

    stage_weights_frag(W1, sW1f, KT1, tid, 256);
    stage_weights_frag(W2, sW2f, KT2, tid, 256);
    __syncthreads();

    __bf16* msgW  = sMsg + w * 16 * MSG_STRIDE;
    __bf16* h1W   = sH1  + w * 16 * H1_STRIDE;
    int*    dstW  = sDst + w * 16;
    float*  distW = sDist + w * 16;

    const float delta = CUTOFF_F / (NBAS - 1);
    const float inv_w = 1.0f / (0.5f * (delta + 0.01f));

    const int nChunks = (numTiles + 7) / 8;
    for (int chunk = blockIdx.x; chunk < nChunks; chunk += gridDim.x) {
        const int tile = chunk * 8 + w;
        const int e0   = tile * 16;

        // ---- prepass: dst index + clamped distance per edge (lanes 0..15)
        if (l < 16) {
            int e = e0 + l; if (e >= NEDGES) e = NEDGES - 1;
            int sN = ei[e];
            int dN = ei[NEDGES + e];
            float dx = pos[dN * 3 + 0] - pos[sN * 3 + 0];
            float dy = pos[dN * 3 + 1] - pos[sN * 3 + 1];
            float dz = pos[dN * 3 + 2] - pos[sN * 3 + 2];
            float dist = sqrtf(dx * dx + dy * dy + dz * dz);
            dstW[l]  = dN;
            distW[l] = fminf(dist, CUTOFF_F);
        }
        __syncthreads();

        // ---- build 16x288 message tile in LDS (bf16)
        for (int m = 0; m < 16; ++m) {
            int e = e0 + m; if (e >= NEDGES) e = NEDGES - 1;
            int sN = ei[e];
            int dN = ei[NEDGES + e];
            const float4 fs = *(const float4*)(nf + (size_t)sN * HID + l * 4);
            const float4 fd = *(const float4*)(nf + (size_t)dN * HID + l * 4);
            __bf16* row = msgW + m * MSG_STRIDE;
            row[l * 4 + 0] = (__bf16)fs.x;
            row[l * 4 + 1] = (__bf16)fs.y;
            row[l * 4 + 2] = (__bf16)fs.z;
            row[l * 4 + 3] = (__bf16)fs.w;
            row[128 + l * 4 + 0] = (__bf16)fd.x;
            row[128 + l * 4 + 1] = (__bf16)fd.y;
            row[128 + l * 4 + 2] = (__bf16)fd.z;
            row[128 + l * 4 + 3] = (__bf16)fd.w;
            float t = (distW[m] - delta * (float)l) * inv_w;   // lane = basis idx
            row[256 + l] = (__bf16)__expf(-0.5f * t * t);
        }
        __syncthreads();

        // ---- layer 1: h1 = silu(msg @ W1 + b1), 9 K-tiles x 8 N-tiles of WMMA
        v16bf a1[KT1];
        {
            const __bf16* arow = msgW + nl * MSG_STRIDE;
#pragma unroll
            for (int kt = 0; kt < KT1; ++kt)
                a1[kt] = load_frag_row(arow, kt * 32 + 8 * hl);
        }
#pragma unroll
        for (int nt = 0; nt < NT; ++nt) {
            float bias = b1[nt * 16 + nl];
            v8f c;
#pragma unroll
            for (int v = 0; v < 8; ++v) c[v] = bias;
#pragma unroll
            for (int kt = 0; kt < KT1; ++kt) {
                v16bf b = load_frag_packed(sW1f + kt * 4096 + nt * 512 + hl * 256 + nl * 16);
                c = __builtin_amdgcn_wmma_f32_16x16x32_bf16(false, a1[kt], false, b,
                                                            (short)0, c, false, false);
            }
            // SiLU, write back in row-major so layer-2 can re-read as A fragments
#pragma unroll
            for (int v = 0; v < 8; ++v) {
                float x = c[v];
                float s = x / (1.0f + __expf(-x));
                h1W[(v + 8 * hl) * H1_STRIDE + nt * 16 + nl] = (__bf16)s;
            }
        }
        __syncthreads();

        // ---- layer 2: h2 = h1 @ W2 + b2, then scatter-add to updates[dst]
        v16bf a2[KT2];
        {
            const __bf16* arow = h1W + nl * H1_STRIDE;
#pragma unroll
            for (int kt = 0; kt < KT2; ++kt)
                a2[kt] = load_frag_row(arow, kt * 32 + 8 * hl);
        }
#pragma unroll
        for (int nt = 0; nt < NT; ++nt) {
            float bias = b2[nt * 16 + nl];
            v8f c;
#pragma unroll
            for (int v = 0; v < 8; ++v) c[v] = bias;
#pragma unroll
            for (int kt = 0; kt < KT2; ++kt) {
                v16bf b = load_frag_packed(sW2f + kt * 4096 + nt * 512 + hl * 256 + nl * 16);
                c = __builtin_amdgcn_wmma_f32_16x16x32_bf16(false, a2[kt], false, b,
                                                            (short)0, c, false, false);
            }
            // C/D layout: VGPR v, lane l -> row M = v + 8*hl, col N = nt*16 + nl
#pragma unroll
            for (int v = 0; v < 8; ++v) {
                int m = v + 8 * hl;
                if (e0 + m < NEDGES) {
                    int node = dstW[m];
                    atomicAdd(&updates[(size_t)node * HID + nt * 16 + nl], c[v]);
                }
            }
        }
        if (l < 16 && (e0 + l) < NEDGES)
            atomicAdd(&degree[dstW[l]], 1.0f);
        __syncthreads();  // protect sMsg/sH1/sDst before next chunk
    }
}

// ---------------------------------------------------------------------------
// Node kernel: out = (updates / max(degree,1)) @ Wo + bo
// ---------------------------------------------------------------------------
__global__ void __launch_bounds__(256, 2)
node_out_kernel(const float* __restrict__ updates, const float* __restrict__ degree,
                const float* __restrict__ Wo, const float* __restrict__ bo,
                float* __restrict__ out) {
    extern __shared__ __align__(16) char smem[];
    __bf16* sWoF = (__bf16*)smem;              // 4*4096 elems
    __bf16* sA   = sWoF + KT2 * 4096;          // 8*16*H1_STRIDE

    const int tid = threadIdx.x;
    const int w   = tid >> 5;
    const int l   = tid & 31;
    const int nl  = l & 15;
    const int hl  = l >> 4;

    stage_weights_frag(Wo, sWoF, KT2, tid, 256);
    __syncthreads();

    __bf16* aW = sA + w * 16 * H1_STRIDE;
    const int n0 = (blockIdx.x * 8 + w) * 16;

    // stage normalized node rows as bf16
    for (int m = 0; m < 16; ++m) {
        int node = n0 + m; if (node >= NNODES) node = NNODES - 1;
        float inv = 1.0f / fmaxf(degree[node], 1.0f);
        float4 u = *(const float4*)(updates + (size_t)node * HID + l * 4);
        __bf16* row = aW + m * H1_STRIDE;
        row[l * 4 + 0] = (__bf16)(u.x * inv);
        row[l * 4 + 1] = (__bf16)(u.y * inv);
        row[l * 4 + 2] = (__bf16)(u.z * inv);
        row[l * 4 + 3] = (__bf16)(u.w * inv);
    }
    __syncthreads();

    v16bf a[KT2];
    {
        const __bf16* arow = aW + nl * H1_STRIDE;
#pragma unroll
        for (int kt = 0; kt < KT2; ++kt)
            a[kt] = load_frag_row(arow, kt * 32 + 8 * hl);
    }
#pragma unroll
    for (int nt = 0; nt < NT; ++nt) {
        float bias = bo[nt * 16 + nl];
        v8f c;
#pragma unroll
        for (int v = 0; v < 8; ++v) c[v] = bias;
#pragma unroll
        for (int kt = 0; kt < KT2; ++kt) {
            v16bf b = load_frag_packed(sWoF + kt * 4096 + nt * 512 + hl * 256 + nl * 16);
            c = __builtin_amdgcn_wmma_f32_16x16x32_bf16(false, a[kt], false, b,
                                                        (short)0, c, false, false);
        }
#pragma unroll
        for (int v = 0; v < 8; ++v) {
            int m = v + 8 * hl;
            int node = n0 + m;
            if (node < NNODES)
                out[(size_t)node * HID + nt * 16 + nl] = c[v];
        }
    }
}

// ---------------------------------------------------------------------------
extern "C" void kernel_launch(void* const* d_in, const int* in_sizes, int n_in,
                              void* d_out, int out_size, void* d_ws, size_t ws_size,
                              hipStream_t stream) {
    (void)in_sizes; (void)n_in; (void)out_size; (void)ws_size;
    const float* nf  = (const float*)d_in[0];
    const float* pos = (const float*)d_in[1];
    const int*   ei  = (const int*)d_in[2];
    const float* W1  = (const float*)d_in[3];
    const float* b1  = (const float*)d_in[4];
    const float* W2  = (const float*)d_in[5];
    const float* b2  = (const float*)d_in[6];
    const float* Wo  = (const float*)d_in[7];
    const float* bo  = (const float*)d_in[8];
    float* out = (float*)d_out;

    float* updates = (float*)d_ws;                       // [NNODES, HID]
    float* degree  = updates + (size_t)NNODES * HID;     // [NNODES]

    // zero the accumulators (harness poisons d_ws; must be deterministic)
    zero_kernel<<<512, 256, 0, stream>>>(updates, (size_t)NNODES * HID + NNODES);

    const int numTiles = (NEDGES + 15) / 16;             // 50000 tiles of 16 edges
    const size_t smemEdge =
        (size_t)(KT1 * 4096 + KT2 * 4096 +
                 8 * 16 * MSG_STRIDE + 8 * 16 * H1_STRIDE) * sizeof(__bf16)
        + 8 * 16 * sizeof(int) + 8 * 16 * sizeof(float);
    edge_mlp_kernel<<<1024, 256, smemEdge, stream>>>(nf, pos, ei, W1, b1, W2, b2,
                                                     updates, degree, numTiles);

    const size_t smemNode =
        (size_t)(KT2 * 4096 + 8 * 16 * H1_STRIDE) * sizeof(__bf16);
    const int nodeBlocks = (NNODES + 127) / 128;
    node_out_kernel<<<nodeBlocks, 256, smemNode, stream>>>(updates, degree, Wo, bo, out);
}
